// self_attention_module_68118181314751
// MI455X (gfx1250) — compile-verified, matrix operand order checked by reading, not run on hardware
//
#include <hip/hip_runtime.h>
#include <hip/hip_bf16.h>

// Problem sizes (fixed by the reference)
#define BD    1024     // B*D
#define CCH   128      // C (EEG channels)
#define TT    1024     // T (time)
#define DQK   256
#define DV    256
#define NCAT  768      // DQK + DQK + DV
#define MROWS 32       // rows per projection m-group (2 WMMA m-tiles)
#define NMG   4        // 128 / MROWS

typedef __attribute__((ext_vector_type(16))) __bf16 v16bf;
typedef __attribute__((ext_vector_type(8)))  float  v8f;
typedef __attribute__((ext_vector_type(4)))  unsigned int u32x4;
typedef __attribute__((ext_vector_type(8)))  int  i32x8;
typedef __attribute__((ext_vector_type(4)))  int  i32x4;

union Frag {          // one WMMA 16x32 bf16 operand: 8 VGPRs / 32 bytes per lane
    v16bf v;
    uint4 q[2];
};

// ---------------------------------------------------------------------------
// Kernel 0: convert Wq|Wk|Wv (f32 [256][1024] each) into concatenated bf16
// Wcat [768][1024] row-major in workspace. 1.5 MB -> lives in L2 afterwards.
// ---------------------------------------------------------------------------
__global__ void prep_weights(const float* __restrict__ Wq,
                             const float* __restrict__ Wk,
                             const float* __restrict__ Wv,
                             __bf16* __restrict__ Wcat)
{
    unsigned i = blockIdx.x * blockDim.x + threadIdx.x;   // 0 .. 768*1024-1
    if (i >= NCAT * TT) return;
    unsigned row = i >> 10;          // /1024
    unsigned col = i & 1023;
    float w;
    if (row < 256)      w = Wq[row * TT + col];
    else if (row < 512) w = Wk[(row - 256) * TT + col];
    else                w = Wv[(row - 512) * TT + col];
    Wcat[i] = (__bf16)w;
}

// ---------------------------------------------------------------------------
// Kernel 1: fused QKV projection + channel attention per (b,d) pair.
// One block (8 waves, 256 threads) per bd. Dynamic LDS layout:
//   xs   : f32  2 x [32][32]  TDM double-buffered x chunk   (8 KB)
//   bias : f32  [768]         bq|bk|bv (async-staged)       (3 KB)
//   Qs   : bf16 [128][256]    Q (row major)                 (64 KB)
//   Ks   : bf16 [128][256]    K (row major)                 (64 KB)
//   Vts  : bf16 [256][128]    V transposed (dv major)       (64 KB)
//   Ps   : bf16 [128][128]    softmax probs (row major)     (32 KB)
// ---------------------------------------------------------------------------
#define LDS_XS    0
#define LDS_BIAS  8192
#define LDS_QS    11264
#define LDS_KS    (11264 + 65536)
#define LDS_VTS   (11264 + 2 * 65536)
#define LDS_PS    (11264 + 3 * 65536)
#define LDS_TOT   (11264 + 3 * 65536 + 32768)   // 240640 bytes

__global__ __launch_bounds__(256, 1)
void attn_fused(const float*  __restrict__ x,      // [BD][128][1024]
                const __bf16* __restrict__ Wcat,   // [768][1024]
                const float*  __restrict__ bq,
                const float*  __restrict__ bk,
                const float*  __restrict__ bv,
                float*        __restrict__ out)    // [BD][128][256]
{
    extern __shared__ char smem[];
    float*  xs   = (float*)(smem + LDS_XS);
    float*  bcat = (float*)(smem + LDS_BIAS);
    __bf16* Qs   = (__bf16*)(smem + LDS_QS);
    __bf16* Ks   = (__bf16*)(smem + LDS_KS);
    __bf16* Vts  = (__bf16*)(smem + LDS_VTS);
    __bf16* Ps   = (__bf16*)(smem + LDS_PS);

    const int bd   = blockIdx.x;
    const int tid  = threadIdx.x;
    const int wave = tid >> 5;       // 0..7
    const int lane = tid & 31;
    const int lr   = lane & 15;      // row/col index inside a 16x16 tile
    const int lh   = lane >> 4;      // which 16-lane half (selects K groups)

    const float* xb = x + (size_t)bd * (CCH * TT);
    // low 32 bits of a flat LDS pointer == LDS byte offset (ISA aperture rule)
    const unsigned xs_lds   = (unsigned)(size_t)(void*)xs;
    const unsigned bias_lds = (unsigned)(size_t)(void*)bcat;

    // --- one-shot async-to-LDS preload of concatenated biases (768 f32) ---
#pragma unroll
    for (int e = 0; e < 3; ++e) {
        const int idx = tid + e * 256;
        const float* gp = (idx < 256) ? (bq + idx)
                        : (idx < 512) ? (bk + (idx - 256))
                                      : (bv + (idx - 512));
        unsigned la = bias_lds + (unsigned)(idx * 4);
        asm volatile("global_load_async_to_lds_b32 %0, %1, off"
                     :: "v"(la), "v"(gp) : "memory");
    }
    asm volatile("s_wait_asynccnt 0" ::: "memory");

    // --- TDM issue helper: load x[mg*32 .. +31][kc*32 .. +31] (f32) to LDS ---
    auto issue_tdm = [&](int mg, int kc, int buf) {
        unsigned long long ga =
            (unsigned long long)(const void*)(xb + (size_t)(mg * MROWS) * TT + kc * 32);
        u32x4 g0;
        g0[0] = 1u;                                   // count=1, user descriptor
        g0[1] = xs_lds + (unsigned)(buf * MROWS * 32 * 4);   // lds_addr (bytes)
        g0[2] = (unsigned)(ga & 0xFFFFFFFFu);         // global_addr[31:0]
        g0[3] = (unsigned)((ga >> 32) & 0x01FFFFFFu)  // global_addr[56:32]
              | (2u << 30);                           // type = 2 ("image")
        i32x8 g1;
        g1[0] = (int)(2u << 16);                      // data_size = 2 (4 bytes)
        g1[1] = (int)(((unsigned)TT & 0xFFFFu) << 16);   // tensor_dim0 = 1024
        g1[2] = (int)(((unsigned)CCH & 0xFFFFu) << 16);  // tensor_dim1 = 128
        g1[3] = (int)(32u << 16);                     // tile_dim0 = 32
        g1[4] = (int)MROWS;                           // tile_dim1 = 32
        g1[5] = (int)TT;                              // tensor_dim0_stride = 1024
        g1[6] = 0;
        g1[7] = 0;
        i32x4 z4 = (i32x4){0, 0, 0, 0};
        i32x8 z8 = (i32x8){0, 0, 0, 0, 0, 0, 0, 0};
        __builtin_amdgcn_tensor_load_to_lds(g0, g1, z4, z4, z8, 0);
    };

    // =======================================================================
    // Phase 1: [Q|K|V] = x @ Wcat^T + bias.  M=128 (4 m-groups of 32 rows),
    // N=768 (wave w owns cols [96w, 96w+96) = 6 tiles), K=1024 (32 chunks).
    // x chunks arrive via double-buffered TDM, overlapped with WMMA.
    // =======================================================================
    if (wave == 0) issue_tdm(0, 0, 0);

    for (int mg = 0; mg < NMG; ++mg) {
        v8f acc[2][6];
#pragma unroll
        for (int tm = 0; tm < 2; ++tm)
#pragma unroll
            for (int t = 0; t < 6; ++t)
                acc[tm][t] = (v8f){0.f,0.f,0.f,0.f,0.f,0.f,0.f,0.f};

        for (int kc = 0; kc < 32; ++kc) {
            const int cur = kc & 1;
            if (wave == 0) __builtin_amdgcn_s_wait_tensorcnt(0);
            __syncthreads();                       // current chunk visible; old reads retired
            if (wave == 0) {                       // prefetch next chunk into other buffer
                int nkc = kc + 1, nmg = mg;
                if (nkc == 32) { nkc = 0; nmg = mg + 1; }
                if (nmg < NMG) issue_tdm(nmg, nkc, cur ^ 1);
            }

            // --- A fragments (two 16x32 bf16 tiles, ISA 16-bit A layout) ---
            const float* xsbuf = xs + cur * (MROWS * 32);
            Frag a[2];
#pragma unroll
            for (int tm = 0; tm < 2; ++tm) {
                const int row = tm * 16 + lr;
#pragma unroll
                for (int j = 0; j < 8; ++j) {
                    a[tm].v[j]     = (__bf16)xsbuf[row * 32 + lh * 8 + j];
                    a[tm].v[j + 8] = (__bf16)xsbuf[row * 32 + 16 + lh * 8 + j];
                }
            }

            // --- 6 B fragments from Wcat rows (contiguous 16B pairs) ---
            const int k0 = kc * 32;
#pragma unroll
            for (int tn = 0; tn < 6; ++tn) {
                const int ncol = wave * 96 + tn * 16 + lr;
                const __bf16* wrow = Wcat + (size_t)ncol * TT + k0 + lh * 8;
                Frag b;
                b.q[0] = *(const uint4*)(wrow);
                b.q[1] = *(const uint4*)(wrow + 16);
                __builtin_prefetch(wrow + 32, 0, 1);   // next kc chunk of this row
                acc[0][tn] = __builtin_amdgcn_wmma_f32_16x16x32_bf16(
                    false, a[0].v, false, b.v, (short)0, acc[0][tn], false, false);
                acc[1][tn] = __builtin_amdgcn_wmma_f32_16x16x32_bf16(
                    false, a[1].v, false, b.v, (short)0, acc[1][tn], false, false);
            }
        }

        // --- epilogue: bias (from LDS), bf16-pack into Qs / Ks / Vts.
        //     Branch on the wave-uniform tile base to avoid exec-mask ladders.
#pragma unroll
        for (int tn = 0; tn < 6; ++tn) {
            const int nb = wave * 96 + tn * 16;       // tile base column (uniform)
            const float bias = bcat[nb + lr];
            if (nb < 256) {
#pragma unroll
                for (int tm = 0; tm < 2; ++tm)
#pragma unroll
                    for (int j = 0; j < 8; ++j) {
                        const int row = mg * MROWS + tm * 16 + lh * 8 + j;
                        Qs[row * 256 + nb + lr] = (__bf16)(acc[tm][tn][j] + bias);
                    }
            } else if (nb < 512) {
#pragma unroll
                for (int tm = 0; tm < 2; ++tm)
#pragma unroll
                    for (int j = 0; j < 8; ++j) {
                        const int row = mg * MROWS + tm * 16 + lh * 8 + j;
                        Ks[row * 256 + (nb - 256) + lr] = (__bf16)(acc[tm][tn][j] + bias);
                    }
            } else {
#pragma unroll
                for (int tm = 0; tm < 2; ++tm)
#pragma unroll
                    for (int j = 0; j < 8; ++j) {
                        const int row = mg * MROWS + tm * 16 + lh * 8 + j;
                        Vts[(nb - 512 + lr) * 128 + row] = (__bf16)(acc[tm][tn][j] + bias);
                    }
            }
        }
    }
    __syncthreads();

    // =======================================================================
    // Phase 2: S = (Q K^T)/sqrt(DQK), softmax rows, store bf16 probs.
    // Wave w owns score rows [16w, 16w+16); 8 n-tiles; K = 256 (8 chunks).
    // =======================================================================
    v8f s[8];
#pragma unroll
    for (int t = 0; t < 8; ++t) s[t] = (v8f){0.f,0.f,0.f,0.f,0.f,0.f,0.f,0.f};

    for (int kc = 0; kc < 8; ++kc) {
        const int k0 = kc * 32;
        Frag a;
        const __bf16* qrow = Qs + (wave * 16 + lr) * 256 + k0 + lh * 8;
        a.q[0] = *(const uint4*)(qrow);
        a.q[1] = *(const uint4*)(qrow + 16);
#pragma unroll
        for (int tn = 0; tn < 8; ++tn) {
            const __bf16* krow = Ks + (tn * 16 + lr) * 256 + k0 + lh * 8;
            Frag b;
            b.q[0] = *(const uint4*)(krow);
            b.q[1] = *(const uint4*)(krow + 16);
            s[tn] = __builtin_amdgcn_wmma_f32_16x16x32_bf16(
                false, a.v, false, b.v, (short)0, s[tn], false, false);
        }
    }

    // register-resident softmax; row (lh,j) lives in 16 lanes of this half
    const float scale = 0.0625f;   // 1/sqrt(256)
#pragma unroll
    for (int j = 0; j < 8; ++j) {
        float vals[8];
        float mx = -3.0e38f;
#pragma unroll
        for (int tn = 0; tn < 8; ++tn) { vals[tn] = s[tn][j] * scale; mx = fmaxf(mx, vals[tn]); }
        for (int msk = 8; msk >= 1; msk >>= 1) mx = fmaxf(mx, __shfl_xor(mx, msk, 32));
        float sum = 0.f;
#pragma unroll
        for (int tn = 0; tn < 8; ++tn) { vals[tn] = __expf(vals[tn] - mx); sum += vals[tn]; }
        for (int msk = 8; msk >= 1; msk >>= 1) sum += __shfl_xor(sum, msk, 32);
        const float inv = 1.0f / sum;
        const int row = wave * 16 + lh * 8 + j;
#pragma unroll
        for (int tn = 0; tn < 8; ++tn)
            Ps[row * 128 + tn * 16 + lr] = (__bf16)(vals[tn] * inv);
    }
    __syncthreads();

    // =======================================================================
    // Phase 3: Z = P @ V.  Wave w owns output rows [16w,16w+16); 256 cols in
    // two halves of 8 tiles; K = 128 (4 chunks). B fragments = rows of V^T.
    // =======================================================================
    for (int nh = 0; nh < 2; ++nh) {
        v8f z[8];
#pragma unroll
        for (int t = 0; t < 8; ++t) z[t] = (v8f){0.f,0.f,0.f,0.f,0.f,0.f,0.f,0.f};

        for (int kc = 0; kc < 4; ++kc) {
            const int k0 = kc * 32;
            Frag a;
            const __bf16* prow = Ps + (wave * 16 + lr) * 128 + k0 + lh * 8;
            a.q[0] = *(const uint4*)(prow);
            a.q[1] = *(const uint4*)(prow + 16);
#pragma unroll
            for (int tn = 0; tn < 8; ++tn) {
                const int ncol = nh * 128 + tn * 16 + lr;
                const __bf16* vrow = Vts + ncol * 128 + k0 + lh * 8;
                Frag b;
                b.q[0] = *(const uint4*)(vrow);
                b.q[1] = *(const uint4*)(vrow + 16);
                z[tn] = __builtin_amdgcn_wmma_f32_16x16x32_bf16(
                    false, a.v, false, b.v, (short)0, z[tn], false, false);
            }
        }
#pragma unroll
        for (int tn = 0; tn < 8; ++tn) {
#pragma unroll
            for (int j = 0; j < 8; ++j) {
                const int row = wave * 16 + lh * 8 + j;
                const int col = nh * 128 + tn * 16 + lr;
                out[((size_t)bd * CCH + row) * DV + col] = z[tn][j];
            }
        }
    }
}

// ---------------------------------------------------------------------------
extern "C" void kernel_launch(void* const* d_in, const int* in_sizes, int n_in,
                              void* d_out, int out_size, void* d_ws, size_t ws_size,
                              hipStream_t stream)
{
    const float* x  = (const float*)d_in[0];
    const float* Wq = (const float*)d_in[1];
    const float* bq = (const float*)d_in[2];
    const float* Wk = (const float*)d_in[3];
    const float* bk = (const float*)d_in[4];
    const float* Wv = (const float*)d_in[5];
    const float* bv = (const float*)d_in[6];

    __bf16* Wcat = (__bf16*)d_ws;   // 768*1024*2 = 1.5 MB of workspace

    prep_weights<<<(NCAT * TT + 255) / 256, 256, 0, stream>>>(Wq, Wk, Wv, Wcat);

    (void)hipFuncSetAttribute((const void*)attn_fused,
                              hipFuncAttributeMaxDynamicSharedMemorySize, LDS_TOT);
    attn_fused<<<BD, 256, LDS_TOT, stream>>>(x, Wcat, bq, bk, bv, (float*)d_out);
}
